// Dynamics_79731772883004
// MI455X (gfx1250) — matrix operand order repaired and checked by reading, not compiled, not run on hardware
//
#include <hip/hip_runtime.h>

typedef _Float16 f16;
typedef _Float16 v16h __attribute__((ext_vector_type(16)));
typedef _Float16 v8h  __attribute__((ext_vector_type(8)));
typedef float    v8f  __attribute__((ext_vector_type(8)));

#define NBLK     1024        // 131072 rows / 128 rows per block
#define EPSV     0.1f
#define ALPHAV   0.05f
#define DEPSV    0.001f

// ---- workspace layout (bytes); identical to first SM_WAVE bytes of LDS ----
#define WS_W1T   0            // [640][128] f16  = 163840  (cols: 128 h | 256 eta | 256 xi), W^T row-major
#define WS_W2HT  163840       // [128][128] f16  = 32768   (h second layer, W^T)
#define WS_EWT   196608       // [16][512]  f16  = 16384   (row0=etaW2 on k<256, row1=xiW2 on k>=256)
#define WS_B1    212992       // 640 f32
#define WS_B2    215552       // 128 f32
#define WS_TOTAL 216064

// ---- shared memory layout ----
#define SM_WAVE  216064
#define WV_Z     0            // [16][128] f32 = 8192
#define WV_PAIR  8192         // [16][32]  f16 = 1024
#define WV_SS    9216         // 16 f32
#define WV_SSH   9280
#define WV_ETA   9344
#define WV_XI    9408
#define WV_BYTES 9472
#define SMEM_BYTES (SM_WAVE + 8*WV_BYTES)   // 291840 < 320KB

__device__ __forceinline__ float half_reduce16(float v){
  v += __shfl_xor(v, 1, 32);
  v += __shfl_xor(v, 2, 32);
  v += __shfl_xor(v, 4, 32);
  v += __shfl_xor(v, 8, 32);
  return v;  // sum over each 16-lane half
}

// B fragment (32x16 f16): lane holds column n=lane&15; halves 0..7 = K k0+kg*8..+7, halves 8..15 = K k0+kg*8+16..+23
__device__ __forceinline__ v16h load_b_frag(const f16* base, int rowstride, int n, int k0, int kg){
  const v8h* p = (const v8h*)(base + n*rowstride + k0 + kg*8);
  v8h lo = p[0];
  v8h hi = p[2];
  return __builtin_shufflevector(lo, hi, 0,1,2,3,4,5,6,7,8,9,10,11,12,13,14,15);
}

// A fragment (16x32 f16) from row-major [16][32] buffer: lane row = lane&15,
// halves 0..7 = K kg*8..+7, halves 8..15 = K kg*8+16..+23
__device__ __forceinline__ v16h load_a_frag(const f16* buf, int ln, int kg){
  const v8h* p0 = (const v8h*)(buf + ln*32 + kg*8);
  const v8h* p1 = (const v8h*)(buf + ln*32 + kg*8 + 16);
  return __builtin_shufflevector(p0[0], p1[0], 0,1,2,3,4,5,6,7,8,9,10,11,12,13,14,15);
}

__device__ __forceinline__ float elu1(float x){
  return x > 0.f ? x : (__expf(x) - 1.f);
}

__global__ void prep_pack(const float* __restrict__ hW1, const float* __restrict__ hb1,
                          const float* __restrict__ hW2, const float* __restrict__ hb2,
                          const float* __restrict__ eW1, const float* __restrict__ eb1,
                          const float* __restrict__ eW2,
                          const float* __restrict__ xW1, const float* __restrict__ xb1,
                          const float* __restrict__ xW2,
                          f16* __restrict__ W1T, f16* __restrict__ W2HT, f16* __restrict__ EWT,
                          float* __restrict__ B1, float* __restrict__ B2)
{
  int i = blockIdx.x*256 + threadIdx.x;
  if (i < 640*128){
    int n = i >> 7, k = i & 127;
    float v;
    if (n < 128)      v = hW1[k*128 + n];
    else if (n < 384) v = eW1[k*256 + (n-128)];
    else              v = xW1[k*256 + (n-384)];
    W1T[i] = (f16)v;
  }
  if (i < 128*128){
    int n = i >> 7, k = i & 127;
    W2HT[i] = (f16)hW2[k*128 + n];
  }
  if (i < 16*512){
    int n = i >> 9, k = i & 511;
    float v = 0.f;
    if (n == 0 && k < 256)  v = eW2[k];
    if (n == 1 && k >= 256) v = xW2[k-256];
    EWT[i] = (f16)v;
  }
  if (i < 640) B1[i] = (i < 128) ? hb1[i] : (i < 384 ? eb1[i-128] : xb1[i-384]);
  if (i < 128) B2[i] = hb2[i];
}

__global__ __launch_bounds__(256) void fused_dyn(
    const float* __restrict__ x,
    const uint4* __restrict__ wsbase,
    const float* __restrict__ etab2p, const float* __restrict__ xib2p,
    const float* __restrict__ rp,
    float* __restrict__ out)
{
  extern __shared__ __align__(16) char smem[];
  f16*   sW1 = (f16*)(smem + WS_W1T);    // [640][128]
  f16*   sW2 = (f16*)(smem + WS_W2HT);   // [128][128]
  f16*   sEW = (f16*)(smem + WS_EWT);    // [16][512]
  float* sB1 = (float*)(smem + WS_B1);
  float* sB2 = (float*)(smem + WS_B2);

  // cooperative weight fill: ws layout == first SM_WAVE bytes of smem
  for (int i = threadIdx.x; i < WS_TOTAL/16; i += 256)
    ((uint4*)smem)[i] = wsbase[i];
  __syncthreads();

  const int wave = threadIdx.x >> 5;
  const int lane = threadIdx.x & 31;
  const int ln   = lane & 15;
  const int kg   = lane >> 4;

  char*  wbase = smem + SM_WAVE + wave*WV_BYTES;
  float* sZ    = (float*)(wbase + WV_Z);
  f16*   sP    = (f16*)(wbase + WV_PAIR);
  float* sSS   = (float*)(wbase + WV_SS);
  float* sSSH  = (float*)(wbase + WV_SSH);
  float* sETA  = (float*)(wbase + WV_ETA);
  float* sXI   = (float*)(wbase + WV_XI);

  const float rv = *rp; const float r2 = rv*rv;
  const float etab2 = *etab2p, xib2 = *xib2p;
  const v8f zero = {0.f,0.f,0.f,0.f,0.f,0.f,0.f,0.f};

  for (int blk = blockIdx.x; blk < NBLK; blk += gridDim.x)
  {
    const int row0 = blk*128 + wave*16;
    const float4* zg4 = (const float4*)(x + (size_t)row0*128);

    // ---- z tile -> LDS; per-row sum(z^2) and head(<96) sum ----
    #pragma unroll
    for (int i = 0; i < 8; ++i){
      int m = i*2 + kg;
      float4 a = zg4[m*32 + ln];
      float4 b = zg4[m*32 + ln + 16];
      ((float4*)(sZ + m*128))[ln]      = a;
      ((float4*)(sZ + m*128))[ln + 16] = b;
      float da = a.x*a.x + a.y*a.y + a.z*a.z + a.w*a.w;
      float db = b.x*b.x + b.y*b.y + b.z*b.z + b.w*b.w;
      float s  = half_reduce16(da + db);
      float sh = half_reduce16(da + (ln < 8 ? db : 0.f));  // cols 64+4*ln head iff ln<8
      if (ln == 0){ sSS[m] = s; sSSH[m] = sh; }
    }

    // ---- z A-fragments (4 K-chunks of 32), f32 -> f16 ----
    v16h zf[4];
    #pragma unroll
    for (int kc = 0; kc < 4; ++kc){
      const float4* p0 = (const float4*)(sZ + ln*128 + kc*32 + kg*8);
      const float4* p1 = (const float4*)(sZ + ln*128 + kc*32 + kg*8 + 16);
      float4 a0 = p0[0], a1 = p0[1];
      float4 b0 = p1[0], b1 = p1[1];
      v8h lo, hi;
      lo[0]=(f16)a0.x; lo[1]=(f16)a0.y; lo[2]=(f16)a0.z; lo[3]=(f16)a0.w;
      lo[4]=(f16)a1.x; lo[5]=(f16)a1.y; lo[6]=(f16)a1.z; lo[7]=(f16)a1.w;
      hi[0]=(f16)b0.x; hi[1]=(f16)b0.y; hi[2]=(f16)b0.z; hi[3]=(f16)b0.w;
      hi[4]=(f16)b1.x; hi[5]=(f16)b1.y; hi[6]=(f16)b1.z; hi[7]=(f16)b1.w;
      zf[kc] = __builtin_shufflevector(lo, hi, 0,1,2,3,4,5,6,7,8,9,10,11,12,13,14,15);
    }

    v8f hacc[8];
    #pragma unroll
    for (int t = 0; t < 8; ++t) hacc[t] = zero;

    // ===== phase 1: p = 0..3  (a1_h production + h = a1_h @ hW2) =====
    for (int p = 0; p < 4; ++p){
      v8f c0 = zero, c1 = zero;
      #pragma unroll
      for (int kc = 0; kc < 4; ++kc){
        v16h bf0 = load_b_frag(sW1, 128, (p*2)*16   + ln, kc*32, kg);
        v16h bf1 = load_b_frag(sW1, 128, (p*2+1)*16 + ln, kc*32, kg);
        c0 = __builtin_amdgcn_wmma_f32_16x16x32_f16(false, zf[kc], false, bf0, (short)0, c0, false, false);
        c1 = __builtin_amdgcn_wmma_f32_16x16x32_f16(false, zf[kc], false, bf1, (short)0, c1, false, false);
      }
      float bb0 = sB1[p*32 + ln];
      float bb1 = sB1[p*32 + 16 + ln];
      #pragma unroll
      for (int j = 0; j < 8; ++j){
        int m = j + 8*kg;
        sP[m*32 + ln]      = (f16)elu1(c0[j] + bb0);
        sP[m*32 + 16 + ln] = (f16)elu1(c1[j] + bb1);
      }
      v16h af = load_a_frag(sP, ln, kg);
      // double-buffered B fragments -> loads overlap WMMA latency
      v16h bcur = load_b_frag(sW2, 128, ln, p*32, kg);
      #pragma unroll
      for (int nt = 0; nt < 8; ++nt){
        v16h bnxt = bcur;
        if (nt < 7) bnxt = load_b_frag(sW2, 128, (nt+1)*16 + ln, p*32, kg);
        hacc[nt] = __builtin_amdgcn_wmma_f32_16x16x32_f16(false, af, false, bcur, (short)0, hacc[nt], false, false);
        bcur = bnxt;
      }
    }

    // ===== phase 2: p = 4..19  ([a1_eta|a1_xi] @ [etaW2;xiW2]) =====
    v8f ex0 = zero, ex1 = zero;
    #pragma unroll 2
    for (int p = 4; p < 20; ++p){
      v8f c0 = zero, c1 = zero;
      #pragma unroll
      for (int kc = 0; kc < 4; ++kc){
        v16h bf0 = load_b_frag(sW1, 128, (p*2)*16   + ln, kc*32, kg);
        v16h bf1 = load_b_frag(sW1, 128, (p*2+1)*16 + ln, kc*32, kg);
        c0 = __builtin_amdgcn_wmma_f32_16x16x32_f16(false, zf[kc], false, bf0, (short)0, c0, false, false);
        c1 = __builtin_amdgcn_wmma_f32_16x16x32_f16(false, zf[kc], false, bf1, (short)0, c1, false, false);
      }
      float bb0 = sB1[p*32 + ln];
      float bb1 = sB1[p*32 + 16 + ln];
      #pragma unroll
      for (int j = 0; j < 8; ++j){
        int m = j + 8*kg;
        sP[m*32 + ln]      = (f16)elu1(c0[j] + bb0);
        sP[m*32 + 16 + ln] = (f16)elu1(c1[j] + bb1);
      }
      v16h af = load_a_frag(sP, ln, kg);
      v16h bf = load_b_frag(sEW, 512, ln, (p-4)*32, kg);
      if (p & 1) ex1 = __builtin_amdgcn_wmma_f32_16x16x32_f16(false, af, false, bf, (short)0, ex1, false, false);
      else       ex0 = __builtin_amdgcn_wmma_f32_16x16x32_f16(false, af, false, bf, (short)0, ex0, false, false);
    }
    v8f exacc = ex0 + ex1;

    // ---- eta/xi scalars (cols 0/1 of ex tile) -> LDS broadcast ----
    #pragma unroll
    for (int j = 0; j < 8; ++j){
      if (ln < 2){
        float v = exacc[j] + (ln == 0 ? etab2 : xib2);
        v = fmaxf(v, 0.f);
        float* dst = (ln == 0) ? sETA : sXI;
        dst[j + 8*kg] = v;
      }
    }

    // ---- add h bias; per-row z·h sums (full & head tiles 0..5) ----
    float szh[8], szhh[8];
    #pragma unroll
    for (int j = 0; j < 8; ++j){ szh[j] = 0.f; szhh[j] = 0.f; }
    #pragma unroll
    for (int t = 0; t < 8; ++t){
      float bb = sB2[t*16 + ln];
      #pragma unroll
      for (int j = 0; j < 8; ++j){
        float hv = hacc[t][j] + bb;
        hacc[t][j] = hv;
        float zv = sZ[(j + 8*kg)*128 + t*16 + ln];
        szh[j] += zv*hv;
        if (t < 6) szhh[j] += zv*hv;
      }
    }
    #pragma unroll
    for (int j = 0; j < 8; ++j){
      szh[j]  = half_reduce16(szh[j]);
      szhh[j] = half_reduce16(szhh[j]);
    }

    // ---- analytic epilogue + store f ----
    #pragma unroll
    for (int j = 0; j < 8; ++j){
      int m = j + 8*kg;
      float ss  = sSS[m];
      float ssh = sSSH[m];
      float y = ss - r2;
      float V  = (y <= 0.f) ? 0.f : (y < EPSV ? y*y*(0.5f/EPSV) : y - 0.5f*EPSV);
      float sp = (y <= 0.f) ? 0.f : (y < EPSV ? y*(1.f/EPSV)    : 1.f);
      float eta = sETA[m], xi = sXI[m];
      float cond = 2.f*sp*szh[j] + ALPHAV*V;
      float c1 = 0.f;
      if (V != 0.f && cond > 0.f){
        float ngV = fmaxf(4.f*sp*sp*ss, 1e-9f);
        c1 = (cond + eta)/ngV * (2.f*sp);
      }
      float c2 = 0.f;
      if (fabsf(y) < DEPSV){
        float ngC = fmaxf(4.f*ssh, 1e-9f);
        c2 = 2.f*(2.f*(szhh[j] - c1*ssh) - xi)/ngC;
      }
      float* orow = out + (size_t)(row0 + m)*128;
      #pragma unroll
      for (int t = 0; t < 8; ++t){
        float zv = sZ[m*128 + t*16 + ln];
        float cc = c1 + (t < 6 ? c2 : 0.f);
        orow[t*16 + ln] = hacc[t][j] - cc*zv;
      }
    }
  }
}

extern "C" void kernel_launch(void* const* d_in, const int* in_sizes, int n_in,
                              void* d_out, int out_size, void* d_ws, size_t ws_size,
                              hipStream_t stream)
{
  (void)in_sizes; (void)n_in; (void)out_size; (void)ws_size;
  const float* x   = (const float*)d_in[1];
  const float* hW1 = (const float*)d_in[2];
  const float* hb1 = (const float*)d_in[3];
  const float* hW2 = (const float*)d_in[4];
  const float* hb2 = (const float*)d_in[5];
  const float* eW1 = (const float*)d_in[6];
  const float* eb1 = (const float*)d_in[7];
  const float* eW2 = (const float*)d_in[8];
  const float* eb2 = (const float*)d_in[9];
  const float* xW1 = (const float*)d_in[10];
  const float* xb1 = (const float*)d_in[11];
  const float* xW2 = (const float*)d_in[12];
  const float* xb2 = (const float*)d_in[13];
  const float* rr  = (const float*)d_in[14];

  char* ws = (char*)d_ws;
  f16*   W1T  = (f16*)(ws + WS_W1T);
  f16*   W2HT = (f16*)(ws + WS_W2HT);
  f16*   EWT  = (f16*)(ws + WS_EWT);
  float* B1   = (float*)(ws + WS_B1);
  float* B2   = (float*)(ws + WS_B2);

  hipFuncSetAttribute(reinterpret_cast<const void*>(&fused_dyn),
                      hipFuncAttributeMaxDynamicSharedMemorySize, SMEM_BYTES);

  prep_pack<<<320, 256, 0, stream>>>(hW1, hb1, hW2, hb2, eW1, eb1, eW2, xW1, xb1, xW2,
                                     W1T, W2HT, EWT, B1, B2);
  fused_dyn<<<256, 256, SMEM_BYTES, stream>>>(x, (const uint4*)ws, eb2, xb2, rr, (float*)d_out);
}
